// ROLLOUT_66451734004406
// MI455X (gfx1250) — matrix operand order repaired and checked by reading, not compiled, not run on hardware
//
#include <hip/hip_runtime.h>
#include <hip/hip_bf16.h>
#include <stddef.h>

// ---------------- problem constants ----------------
#define BATCH   64
#define SEQLEN  128
#define VOCAB   32000
#define EMB     256
#define HID     512
#define NTILES_V (VOCAB / 16)   // 2000
#define KT_EMB  (EMB / 32)      // 8
#define KT_HID  (HID / 32)      // 16

// ---------------- WMMA types (CDNA5 / gfx1250, wave32) ----------------
typedef __attribute__((ext_vector_type(16))) __bf16 v16bf;
typedef __attribute__((ext_vector_type(8)))  float  v8f;

// ---------------- helpers ----------------
__device__ __forceinline__ unsigned short f2bf(float f) {
    unsigned int u = __float_as_uint(f);
    unsigned int r = u + 0x7FFFu + ((u >> 16) & 1u);   // round-to-nearest-even
    return (unsigned short)(r >> 16);
}

__device__ __forceinline__ unsigned int fkey(float f) {
    unsigned int b = __float_as_uint(f);
    return (b & 0x80000000u) ? ~b : (b | 0x80000000u);
}

__device__ __forceinline__ float u01(unsigned int s) {
    s ^= s >> 17; s *= 0xED5AD4BBu;
    s ^= s >> 11; s *= 0xAC4C1B51u;
    s ^= s >> 15; s *= 0x31848BABu;
    s ^= s >> 14;
    return ((s >> 8) + 0.5f) * (1.0f / 16777216.0f);  // strictly in (0,1)
}

__device__ __forceinline__ float sigmoidf(float z) {
    return 1.0f / (1.0f + __expf(-z));
}

// A-fragment global layout (bf16, kt-major):
//   dst[((kt*4 + mt)*32 + lane)*16 + j]
//   row = mt*16 + (lane&15)
//   k   = kt*32 + ((lane<16)?0:8) + ((j<8)? j : j+8)
// k runs CONSECUTIVELY within each j-half (j<8 and j>=8), so each half is an
// 8-wide contiguous fp32 read and one 16-byte bf16 store.

// ---------------- weight packing: fp32 [K][N] -> bf16 B-fragment tiles ----------------
// Tile (kt, nt) at ((nt<<ktShift)+kt)*512 ; lane 0-15: K=kt*32+((j<8)?j:j+8), N=nt*16+lane
//                                          lane16-31: K=kt*32+8+((j<8)?j:j+8), N=nt*16+(lane-16)
// 8 elements per thread (one j-half), K consecutive -> strided gather of W, uint4 store.
__global__ __launch_bounds__(256) void pack_b_kernel(const float* __restrict__ W,
                                                     unsigned short* __restrict__ dst,
                                                     int N, int ktShift) {
    int Kt = 1 << ktShift;
    int K  = Kt << 5;
    int totalG = (K * N) >> 3;                 // groups of 8 bf16
    int stride = gridDim.x * blockDim.x;
    for (int g = blockIdx.x * blockDim.x + threadIdx.x; g < totalG; g += stride) {
        int tile   = g >> 6;
        int within = g & 63;
        int lane = within >> 1;
        int j0   = (within & 1) << 3;          // 0 or 8
        int kt = tile & (Kt - 1);
        int nt = tile >> ktShift;
        int k0 = (kt << 5) + ((lane < 16) ? 0 : 8) + (j0 << 1);
        int n  = (nt << 4) + (lane & 15);
        const float* src = W + (size_t)k0 * N + n;   // stride N between k's
        unsigned int p[4];
#pragma unroll
        for (int q = 0; q < 4; ++q) {
            unsigned int lo = f2bf(src[(size_t)(2 * q)     * N]);
            unsigned int hi = f2bf(src[(size_t)(2 * q + 1) * N]);
            p[q] = lo | (hi << 16);
        }
        *(uint4*)(dst + ((size_t)g << 3)) = make_uint4(p[0], p[1], p[2], p[3]);
    }
}

// ---------------- pack fp32 h [64][512] -> bf16 A-fragments (4096 uint4 groups) ----------------
__global__ __launch_bounds__(256) void pack_h_kernel(const float* __restrict__ h,
                                                     unsigned short* __restrict__ hfrag) {
    int g = blockIdx.x * blockDim.x + threadIdx.x;     // 0..4095
    if (g >= (4 * KT_HID * 512) / 8) return;
    int tile   = g >> 6;                 // kt*4 + mt
    int within = g & 63;
    int lane = within >> 1;
    int jh   = within & 1;
    int mt = tile & 3;
    int kt = tile >> 2;                  // 0..15
    int row   = mt * 16 + (lane & 15);
    int abase = (lane < 16) ? 0 : 8;
    int k0 = kt * 32 + abase + jh * 16;  // 8 consecutive k's
    const float* src = h + (size_t)row * HID + k0;
    unsigned int p[4];
#pragma unroll
    for (int q = 0; q < 4; ++q)
        p[q] = (unsigned int)f2bf(src[2 * q]) | ((unsigned int)f2bf(src[2 * q + 1]) << 16);
    *(uint4*)(hfrag + ((size_t)g << 3)) = make_uint4(p[0], p[1], p[2], p[3]);
}

// ---------------- init: c=0, best=0, hfrag0=0, xfrag = frag(emb[START_TOKEN=0]) ----------------
__global__ __launch_bounds__(256) void init_state_kernel(unsigned short* __restrict__ xfrag,
                                                         unsigned short* __restrict__ hfrag0,
                                                         float* __restrict__ cst,
                                                         unsigned long long* __restrict__ best,
                                                         const float* __restrict__ gemb) {
    int i = blockIdx.x * blockDim.x + threadIdx.x;
    int stride = gridDim.x * blockDim.x;
    for (int k = i; k < BATCH * HID; k += stride) cst[k] = 0.0f;
    for (int k = i; k < 4 * KT_HID * 512; k += stride) hfrag0[k] = 0;
    for (int idx = i; idx < 4 * KT_EMB * 512; idx += stride) {
        int j    = idx & 15;
        int lane = (idx >> 4) & 31;
        int kt   = (idx >> 9) >> 2;     // tile = kt*4+mt; value depends only on k
        int abase = (lane < 16) ? 0 : 8;
        int jj    = (j < 8) ? j : (j + 8);
        int k = kt * 32 + abase + jj;
        xfrag[idx] = f2bf(gemb[k]);     // row START_TOKEN = 0
    }
    for (int k = i; k < BATCH; k += stride) best[k] = 0ull;
}

// ---------------- LSTM gate kernel ----------------
// 128 waves (grid 16 x 256). Wave -> (mt,nt). Pure fragment loads + WMMA.
__global__ __launch_bounds__(256) void lstm_gates_kernel(
        const unsigned short* __restrict__ xfrag,   // 8kt x 4mt x 512
        const unsigned short* __restrict__ hfrag,   // 16kt x 4mt x 512
        float* __restrict__ hout, float* __restrict__ cst,
        const unsigned short* __restrict__ Wp,      // 4 gates, 256x512 packed
        const unsigned short* __restrict__ Up,      // 4 gates, 512x512 packed
        const float* __restrict__ bi_, const float* __restrict__ bf_,
        const float* __restrict__ bo_, const float* __restrict__ bc_,
        unsigned long long* __restrict__ best) {
    int gtid = blockIdx.x * blockDim.x + threadIdx.x;
    if (gtid < BATCH) best[gtid] = 0ull;      // reset argmax slots for this step

    int lane = threadIdx.x & 31;
    int wave = gtid >> 5;                     // 0..127
    int mt = wave & 3;
    int nt = wave >> 2;                       // 0..31

    v8f acc[4] = {};

    // x @ W{i,f,o,c}
#pragma unroll
    for (int kt = 0; kt < KT_EMB; ++kt) {
        const v16bf af = *(const v16bf*)(xfrag + (((size_t)(kt * 4 + mt)) * 32 + lane) * 16);
#pragma unroll
        for (int g = 0; g < 4; ++g) {
            const v16bf bfr = *(const v16bf*)(Wp + (size_t)g * (EMB * HID)
                                                 + (((size_t)nt * KT_EMB + kt) << 9)
                                                 + lane * 16);
            acc[g] = __builtin_amdgcn_wmma_f32_16x16x32_bf16(
                         false, af, false, bfr, (short)0, acc[g], false, false);
        }
    }
    // h @ U{i,f,o,c}
#pragma unroll
    for (int kt = 0; kt < KT_HID; ++kt) {
        const v16bf af = *(const v16bf*)(hfrag + (((size_t)(kt * 4 + mt)) * 32 + lane) * 16);
#pragma unroll
        for (int g = 0; g < 4; ++g) {
            const v16bf bfr = *(const v16bf*)(Up + (size_t)g * (HID * HID)
                                                 + (((size_t)nt * KT_HID + kt) << 9)
                                                 + lane * 16);
            acc[g] = __builtin_amdgcn_wmma_f32_16x16x32_bf16(
                         false, af, false, bfr, (short)0, acc[g], false, false);
        }
    }

    // activations + state update (C/D layout: VGPR v, lane l -> M=v+8*(l>=16), N=l&15)
    int col = nt * 16 + (lane & 15);
    float bI = bi_[col], bF = bf_[col], bO = bo_[col], bC = bc_[col];
    int rbase = mt * 16 + ((lane >= 16) ? 8 : 0);
#pragma unroll
    for (int v = 0; v < 8; ++v) {
        int row = rbase + v;
        float ig = sigmoidf(acc[0][v] + bI);
        float fg = sigmoidf(acc[1][v] + bF);
        float og = sigmoidf(acc[2][v] + bO);
        float ch = tanhf(acc[3][v] + bC);
        size_t off = (size_t)row * HID + col;
        float cn = fg * cst[off] + ig * ch;
        cst[off]  = cn;
        hout[off] = og * tanhf(cn);
    }
}

// ---------------- logits GEMM + Gumbel-max sampling ----------------
// 250 blocks x 256 threads = 2000 waves, one vocab N-tile each.
// Prepacked h fragments staged to LDS in two 32KB phases (raw uint4 copy).
__global__ __launch_bounds__(256) void logits_sample_kernel(
        const unsigned short* __restrict__ hfrag,   // 16kt x 4mt x 512, kt-major
        const unsigned short* __restrict__ Wop,
        const float* __restrict__ bo, unsigned long long* __restrict__ best,
        int step) {
    __shared__ uint4 lds4[2048];                    // 32 KB per phase
    unsigned short* lds = (unsigned short*)lds4;

    int lane = threadIdx.x & 31;
    int widx = threadIdx.x >> 5;
    int nt = blockIdx.x * 8 + widx;                 // 0..1999

    v8f acc[4] = {};

    for (int phase = 0; phase < 2; ++phase) {
        __syncthreads();
        const uint4* src = (const uint4*)(hfrag + phase * 16384);
        for (int i = threadIdx.x; i < 2048; i += 256) lds4[i] = src[i];
        __syncthreads();

#pragma unroll
        for (int kt = 0; kt < 8; ++kt) {
            int gkt = phase * 8 + kt;
            const unsigned short* bptr = Wop + (((size_t)nt * KT_HID + gkt) << 9) + lane * 16;
            if (gkt + 1 < KT_HID)
                __builtin_prefetch(Wop + (((size_t)nt * KT_HID + gkt + 1) << 9) + lane * 16, 0, 1);
            const v16bf bfr = *(const v16bf*)bptr;
#pragma unroll
            for (int m = 0; m < 4; ++m) {
                const v16bf af = *(const v16bf*)(&lds[((kt * 4 + m) * 32 + lane) * 16]);
                acc[m] = __builtin_amdgcn_wmma_f32_16x16x32_bf16(
                             false, af, false, bfr, (short)0, acc[m], false, false);
            }
        }
    }

    // epilogue: bias + Gumbel noise, wave-level max-reduce over the 16 lanes
    // sharing each row, then one atomicMax per (row, half) group.
    int col = nt * 16 + (lane & 15);
    float bias = bo[col];
#pragma unroll
    for (int m = 0; m < 4; ++m) {
        int rbase = m * 16 + ((lane >= 16) ? 8 : 0);
#pragma unroll
        for (int v = 0; v < 8; ++v) {
            int row = rbase + v;
            unsigned int seed = (unsigned int)(step * 73856093)
                              ^ (unsigned int)(row * 19349663)
                              ^ (unsigned int)(col * 83492791);
            float g   = -__logf(-__logf(u01(seed)));
            float val = acc[m][v] + bias + g;
            unsigned long long key =
                ((unsigned long long)fkey(val) << 32) | (unsigned int)col;
#pragma unroll
            for (int d = 1; d < 16; d <<= 1) {
                unsigned long long o = __shfl_xor(key, d, 32);
                key = (o > key) ? o : key;
            }
            if ((lane & 15) == 0) atomicMax(&best[row], key);
        }
    }
}

// ---------------- token select + embedding gather (writes x fragments) ----------------
// One wave per batch row b. Fragment slots with row==b:
//   mt=b>>4, lane in {b&15, 16+(b&15)}, kt=0..7, two j-halves -> 32 uint4 groups.
__global__ __launch_bounds__(32) void sample_embed_kernel(
        const unsigned long long* __restrict__ best,
        const int* __restrict__ input_x, const int* __restrict__ given_num,
        const float* __restrict__ gemb, unsigned short* __restrict__ xfrag,
        int* __restrict__ out_tok, int step) {
    int b = blockIdx.x;
    __shared__ int tok_s;
    if (threadIdx.x == 0) {
        int sampled = (int)(best[b] & 0xFFFFFFFFull);
        int tok = (step < *given_num) ? input_x[b * SEQLEN + step] : sampled;
        tok_s = tok;
        out_tok[b * SEQLEN + step] = tok;
    }
    __syncthreads();
    int tok = tok_s;

    int lane_hi = threadIdx.x >> 4;          // 0: lanes 0-15 half, 1: lanes 16-31 half
    int kt      = (threadIdx.x >> 1) & 7;
    int jh      = threadIdx.x & 1;
    int mt   = b >> 4;
    int lane = lane_hi * 16 + (b & 15);
    int k0 = kt * 32 + (lane_hi ? 8 : 0) + jh * 16;  // 8 consecutive k's
    const float* src = gemb + (size_t)tok * EMB + k0;
    unsigned int p[4];
#pragma unroll
    for (int q = 0; q < 4; ++q)
        p[q] = (unsigned int)f2bf(src[2 * q]) | ((unsigned int)f2bf(src[2 * q + 1]) << 16);
    size_t g = (((size_t)(kt * 4 + mt)) * 32 + lane) * 2 + jh;
    *(uint4*)(xfrag + (g << 3)) = make_uint4(p[0], p[1], p[2], p[3]);
}

// ---------------- workspace layout (bytes) ----------------
#define OFF_XFRAG ((size_t)0)                              // 16384 bf16 = 32768 B
#define OFF_HFA   (OFF_XFRAG + (size_t)4*KT_EMB*512*2)     // 32768 bf16 = 65536 B
#define OFF_HFB   (OFF_HFA   + (size_t)4*KT_HID*512*2)
#define OFF_HTMP  (OFF_HFB   + (size_t)4*KT_HID*512*2)     // 64*512 f32
#define OFF_C     (OFF_HTMP  + (size_t)BATCH*HID*4)
#define OFF_BEST  (OFF_C     + (size_t)BATCH*HID*4)        // 64 u64
#define OFF_WP    (OFF_BEST  + 512)                        // 4 * 256*512 bf16
#define OFF_UP    (OFF_WP    + (size_t)4*EMB*HID*2)        // 4 * 512*512 bf16
#define OFF_WOP   (OFF_UP    + (size_t)4*HID*HID*2)        // 512*32000 bf16

extern "C" void kernel_launch(void* const* d_in, const int* in_sizes, int n_in,
                              void* d_out, int out_size, void* d_ws, size_t ws_size,
                              hipStream_t stream) {
    const int*   input_x   = (const int*)  d_in[0];
    const int*   given_num = (const int*)  d_in[1];
    const float* gemb      = (const float*)d_in[2];
    const float* Wi  = (const float*)d_in[3];
    const float* Ui  = (const float*)d_in[4];
    const float* bi  = (const float*)d_in[5];
    const float* Wf  = (const float*)d_in[6];
    const float* Uf  = (const float*)d_in[7];
    const float* bfv = (const float*)d_in[8];
    const float* Wog = (const float*)d_in[9];
    const float* Uog = (const float*)d_in[10];
    const float* bog = (const float*)d_in[11];
    const float* Wc  = (const float*)d_in[12];
    const float* Uc  = (const float*)d_in[13];
    const float* bc  = (const float*)d_in[14];
    const float* Wo  = (const float*)d_in[15];
    const float* bo  = (const float*)d_in[16];

    char* ws = (char*)d_ws;
    unsigned short* xfrag = (unsigned short*)(ws + OFF_XFRAG);
    unsigned short* hfA   = (unsigned short*)(ws + OFF_HFA);
    unsigned short* hfB   = (unsigned short*)(ws + OFF_HFB);
    float* htmp = (float*)(ws + OFF_HTMP);
    float* c    = (float*)(ws + OFF_C);
    unsigned long long* best = (unsigned long long*)(ws + OFF_BEST);
    unsigned short* Wp  = (unsigned short*)(ws + OFF_WP);
    unsigned short* Up  = (unsigned short*)(ws + OFF_UP);
    unsigned short* Wop = (unsigned short*)(ws + OFF_WOP);

    // ---- one-time (per call) weight packing to bf16 B-fragment layout ----
    pack_b_kernel<<<64,   256, 0, stream>>>(Wi,  Wp + 0*(size_t)EMB*HID, HID,   3);
    pack_b_kernel<<<64,   256, 0, stream>>>(Wf,  Wp + 1*(size_t)EMB*HID, HID,   3);
    pack_b_kernel<<<64,   256, 0, stream>>>(Wog, Wp + 2*(size_t)EMB*HID, HID,   3);
    pack_b_kernel<<<64,   256, 0, stream>>>(Wc,  Wp + 3*(size_t)EMB*HID, HID,   3);
    pack_b_kernel<<<128,  256, 0, stream>>>(Ui,  Up + 0*(size_t)HID*HID, HID,   4);
    pack_b_kernel<<<128,  256, 0, stream>>>(Uf,  Up + 1*(size_t)HID*HID, HID,   4);
    pack_b_kernel<<<128,  256, 0, stream>>>(Uog, Up + 2*(size_t)HID*HID, HID,   4);
    pack_b_kernel<<<128,  256, 0, stream>>>(Uc,  Up + 3*(size_t)HID*HID, HID,   4);
    pack_b_kernel<<<4096, 256, 0, stream>>>(Wo,  Wop, VOCAB, 4);

    init_state_kernel<<<256, 256, 0, stream>>>(xfrag, hfA, c, best, gemb);

    int* out_tok = (int*)d_out;
    for (int t = 0; t < SEQLEN; ++t) {
        const unsigned short* hin  = (t & 1) ? hfB : hfA;
        unsigned short*       hnew = (t & 1) ? hfA : hfB;
        lstm_gates_kernel<<<16, 256, 0, stream>>>(xfrag, hin, htmp, c, Wp, Up,
                                                  bi, bfv, bog, bc, best);
        pack_h_kernel<<<16, 256, 0, stream>>>(htmp, hnew);
        logits_sample_kernel<<<NTILES_V / 8, 256, 0, stream>>>(hnew, Wop, bo, best, t);
        sample_embed_kernel<<<BATCH, 32, 0, stream>>>(best, input_x, given_num,
                                                      gemb, xfrag, out_tok, t);
    }
}
